// QuantLSTMModel_81381040325288
// MI455X (gfx1250) — matrix operand order repaired and checked by reading, not compiled, not run on hardware
//
#include <hip/hip_runtime.h>
#include <hip/hip_bf16.h>

typedef __attribute__((ext_vector_type(16))) _Float16 v16h;
typedef __attribute__((ext_vector_type(8)))  float    v8f;

#define B_SZ 512
#define T_SZ 2048
#define I_SZ 10
#define H_SZ 20
#define G_SZ 80                 // 4*H gate columns
#define QMINF (-128.0f)
#define QMAXF (127.0f)
#define ACC_SCALE 0.125f        // 2^-3
#define ACC_INV   8.0f
#define ACT_SCALE 0.0078125f    // 2^-7 (both sigmoid and tanh quantizers)
#define ACT_INV   128.0f
#define LOG2E     1.4426950408889634f

__device__ __forceinline__ float fq(float v, float s, float inv_s) {
    float q = rintf(v * inv_s);
    q = fminf(fmaxf(q, QMINF), QMAXF);
    return q * s;
}

// Branch-free activations on the hardware transcendental units (v_exp_f32 /
// v_rcp_f32). Error ~1 ulp, far below the int8 quantizer step (2^-7) applied
// immediately after, and with correct saturation at +-inf.
__device__ __forceinline__ float fast_sigmoid(float x) {
    float e = __builtin_amdgcn_exp2f(-LOG2E * x);      // e^-x
    return __builtin_amdgcn_rcpf(1.0f + e);            // 1/(1+e^-x)
}
__device__ __forceinline__ float fast_tanh(float x) {
    float e2 = __builtin_amdgcn_exp2f((2.0f * LOG2E) * x);  // e^(2x)
    return 1.0f - 2.0f * __builtin_amdgcn_rcpf(e2 + 1.0f);  // 1 - 2/(e^2x+1)
}

// One wave (32 threads) per block; each wave owns a tile of 16 batch rows and
// runs the full T=2048 recurrence for them. 32 blocks cover B=512.
__global__ __launch_bounds__(32) void qlstm_wmma_kernel(
    const float* __restrict__ x,    // [B, T, I]
    const float* __restrict__ h0,   // [1, B, H]
    const float* __restrict__ c0,   // [1, B, H]
    const float* __restrict__ w_ih, // [4H, I]
    const float* __restrict__ w_hh, // [4H, H]
    float* __restrict__ out)        // [B, T, H]
{
    __shared__ float gbuf[16 * G_SZ];  // gate tile, D layout spill (16 rows x 80 cols)
    __shared__ float hbuf[16 * H_SZ];  // recurrent h state (16 rows x 20 cols)

    const int  lane  = threadIdx.x & 31;
    const int  row   = lane & 15;       // batch row within tile / N col within tile
    const bool hi    = lane >= 16;
    const int  bBase = blockIdx.x * 16;

    // ---- shared per-tensor power-of-two weight scale (max over both tensors) ----
    float m = 0.f;
    for (int idx = lane; idx < G_SZ * I_SZ; idx += 32) m = fmaxf(m, fabsf(w_ih[idx]));
    for (int idx = lane; idx < G_SZ * H_SZ; idx += 32) m = fmaxf(m, fabsf(w_hh[idx]));
    #pragma unroll
    for (int off = 16; off >= 1; off >>= 1) m = fmaxf(m, __shfl_xor(m, off, 32));
    const float s_w     = exp2f(ceilf(log2f(m / QMAXF)));
    const float inv_s_w = 1.0f / s_w;

    // ---- B fragments: W_cat[k][n], K = [w_ih row(10) | w_hh row(20) | pad(2)] ----
    // B 32x16 f16 layout: lane n(0..15) holds column n with K={0..7, 16..23},
    // lane n+16 holds column n with K={8..15, 24..31}. Quantized weights are exact in f16.
    v16h bfrag[5];
    {
        const int kb = hi ? 8 : 0;
        #pragma unroll
        for (int td = 0; td < 5; ++td) {
            const int n = td * 16 + row;
            #pragma unroll
            for (int j = 0; j < 16; ++j) {
                const int k = (j < 8) ? (kb + j) : (16 + kb + (j - 8));
                float w;
                if (k < I_SZ)             w = w_ih[n * I_SZ + k];
                else if (k < I_SZ + H_SZ) w = w_hh[n * H_SZ + (k - I_SZ)];
                else                      w = 0.f;
                bfrag[td][j] = (_Float16)fq(w, s_w, inv_s_w);
            }
        }
    }

    // ---- init recurrent state: c in registers (10 elems/lane), h in LDS ----
    float c_st[10];
    #pragma unroll
    for (int idx = 0; idx < 10; ++idx) {
        const int e = lane + idx * 32;          // e < 320 = 16*20
        const int r = e / H_SZ, col = e % H_SZ;
        c_st[idx] = c0[(bBase + r) * H_SZ + col];
        hbuf[r * H_SZ + col] = h0[(bBase + r) * H_SZ + col];
    }
    asm volatile("s_wait_dscnt 0" ::: "memory");

    const float* xrow = x + (size_t)(bBase + row) * T_SZ * I_SZ;

    for (int t = 0; t < T_SZ; ++t) {
        // ---- A fragment (16x32 f16): rows = batch, K = [x_t(10) | h(20) | 0 0]
        // lanes 0-15: K = {0..7, 16..23};  lanes 16-31: K = {8..15, 24..31}
        const float* xp = xrow + (size_t)t * I_SZ;
        v16h a;
        if (!hi) {
            #pragma unroll
            for (int j = 0; j < 8; ++j) a[j]     = (_Float16)xp[j];                       // K=0..7  -> x[0..7]
            #pragma unroll
            for (int j = 0; j < 8; ++j) a[8 + j] = (_Float16)hbuf[row * H_SZ + 6 + j];    // K=16..23 -> h[6..13]
        } else {
            a[0] = (_Float16)xp[8];                                                       // K=8,9 -> x[8..9]
            a[1] = (_Float16)xp[9];
            #pragma unroll
            for (int j = 0; j < 6; ++j) a[2 + j] = (_Float16)hbuf[row * H_SZ + j];        // K=10..15 -> h[0..5]
            #pragma unroll
            for (int j = 0; j < 6; ++j) a[8 + j] = (_Float16)hbuf[row * H_SZ + 14 + j];   // K=24..29 -> h[14..19]
            a[14] = (_Float16)0.f;                                                        // K=30,31 pad
            a[15] = (_Float16)0.f;
        }

        // ---- gates[16x80] = A x W^T  (5 WMMA tiles, f32 accumulate) ----
        v8f cz = {};
        v8f acc[5];
        #pragma unroll
        for (int td = 0; td < 5; ++td) {
            acc[td] = __builtin_amdgcn_wmma_f32_16x16x32_f16(
                false, a, false, bfrag[td], (short)0, cz, false, false);
        }

        // ---- spill D tiles to per-wave LDS (wave-synchronous layout exchange)
        {
            const int mofs = hi ? 8 : 0;
            #pragma unroll
            for (int td = 0; td < 5; ++td) {
                #pragma unroll
                for (int r = 0; r < 8; ++r) {
                    gbuf[(mofs + r) * G_SZ + td * 16 + row] = acc[td][r];
                }
            }
        }
        asm volatile("s_wait_dscnt 0" ::: "memory");

        // ---- elementwise LSTM cell update, 10 (row,col) pairs per lane ----
        #pragma unroll
        for (int idx = 0; idx < 10; ++idx) {
            const int e = lane + idx * 32;
            const int r = e / H_SZ, col = e % H_SZ;
            const float* gr = gbuf + r * G_SZ + col;
            float gi = fq(gr[0],        ACC_SCALE, ACC_INV);
            float gf = fq(gr[H_SZ],     ACC_SCALE, ACC_INV);
            float gg = fq(gr[2 * H_SZ], ACC_SCALE, ACC_INV);
            float go = fq(gr[3 * H_SZ], ACC_SCALE, ACC_INV);
            gi = fq(fast_sigmoid(gi), ACT_SCALE, ACT_INV);
            gf = fq(fast_sigmoid(gf), ACT_SCALE, ACT_INV);
            go = fq(fast_sigmoid(go), ACT_SCALE, ACT_INV);
            gg = fq(fast_tanh(gg),    ACT_SCALE, ACT_INV);
            const float c = gf * c_st[idx] + gi * gg;
            c_st[idx] = c;
            const float h = go * fq(fast_tanh(c), ACT_SCALE, ACT_INV);
            hbuf[r * H_SZ + col] = h;
            out[((size_t)(bBase + r) * T_SZ + t) * H_SZ + col] = h;
        }
        asm volatile("s_wait_dscnt 0" ::: "memory");  // hbuf visible before next A build
    }
}

extern "C" void kernel_launch(void* const* d_in, const int* in_sizes, int n_in,
                              void* d_out, int out_size, void* d_ws, size_t ws_size,
                              hipStream_t stream) {
    (void)in_sizes; (void)n_in; (void)d_ws; (void)ws_size; (void)out_size;
    const float* x    = (const float*)d_in[0];
    const float* h0   = (const float*)d_in[1];
    const float* c0   = (const float*)d_in[2];
    const float* w_ih = (const float*)d_in[3];
    const float* w_hh = (const float*)d_in[4];
    float* out = (float*)d_out;

    qlstm_wmma_kernel<<<dim3(B_SZ / 16), dim3(32), 0, stream>>>(
        x, h0, c0, w_ih, w_hh, out);
}
